// RLTokenTransformerBlock_81295140979091
// MI455X (gfx1250) — matrix operand order repaired and checked by reading, not compiled, not run on hardware
//
#include <hip/hip_runtime.h>
#include <hip/hip_bf16.h>

// ---------------------------------------------------------------------------
// Types for CDNA5 WMMA (wave32)
// ---------------------------------------------------------------------------
typedef __attribute__((ext_vector_type(16))) __bf16 v16bf;
typedef __attribute__((ext_vector_type(8)))  float  v8f;

union AF {
    v16bf v;
    uint4 q[2];
};

#define NHEAD 16
#define HDIM  64

// ---------------------------------------------------------------------------
// Weight prepack: fp32 [K,N] row-major -> bf16 WMMA B-fragments.
// Fragment index f = kt*(N/16)+nt; per fragment 32 lanes x 16 bf16.
// Lane l: n = nt*16 + (l&15), k = kt*32 + 16*(l>>4) + e   (e = 0..15)
// ---------------------------------------------------------------------------
__global__ void pack_w_kernel(const float* __restrict__ W, __bf16* __restrict__ out,
                              int K, int N) {
    int tid = blockIdx.x * blockDim.x + threadIdx.x;
    int total = (K / 32) * (N / 16) * 32;
    if (tid >= total) return;
    int lane = tid & 31;
    int frag = tid >> 5;
    int ntiles = N >> 4;
    int nt = frag % ntiles;
    int kt = frag / ntiles;
    int n  = nt * 16 + (lane & 15);
    int kb = kt * 32 + (lane >> 4) * 16;
    __bf16* dst = out + (size_t)frag * 512 + (size_t)lane * 16;
    const float* src = W + (size_t)kb * N + n;
#pragma unroll
    for (int e = 0; e < 16; ++e)
        dst[e] = (__bf16)src[(size_t)e * N];
}

// ---------------------------------------------------------------------------
// RMSNorm: fp32 [T,D] -> bf16 [T,D].  One block (256 thr) per token row.
// ---------------------------------------------------------------------------
__global__ void rmsnorm_kernel(const float* __restrict__ x, const float* __restrict__ gamma,
                               __bf16* __restrict__ out, int D) {
    int t = blockIdx.x;
    const float* row = x + (size_t)t * D;
    float ss = 0.f;
    for (int i = threadIdx.x; i < D; i += blockDim.x) {
        float v = row[i];
        ss += v * v;
    }
    __shared__ float red[256];
    red[threadIdx.x] = ss;
    __syncthreads();
    for (int off = 128; off > 0; off >>= 1) {
        if ((int)threadIdx.x < off) red[threadIdx.x] += red[threadIdx.x + off];
        __syncthreads();
    }
    float scale = rsqrtf(red[0] / (float)D + 1e-6f);
    for (int i = threadIdx.x; i < D; i += blockDim.x)
        out[(size_t)t * D + i] = (__bf16)(row[i] * scale * gamma[i]);
}

// ---------------------------------------------------------------------------
// WMMA GEMM: C[T,N] = A[T,K](bf16) @ Bpacked + bias, epilogue by MODE.
//   MODE 0: bf16 out, row-major [T,N]
//   MODE 1: bf16 out transposed per head -> [b, nh, hd, S]   (for V)
//   MODE 2: fp32 out = acc + bias + res[T,N]                 (residual adds)
// One wave computes a 16(M) x 64(N) strip; software-pipelined with TWO NAMED
// buffer sets (a0/b0, a1/b1) and compile-time indices only, so everything
// stays in VGPRs (runtime-indexed arrays spill to scratch).
// Requires kloops (=K/32) even: K is 1024 or 4096 here.
// ---------------------------------------------------------------------------
template <int MODE>
__global__ void gemm_kernel(const __bf16* __restrict__ A, const __bf16* __restrict__ Bp,
                            const float* __restrict__ bias, const float* __restrict__ res,
                            void* __restrict__ out, int T, int N, int K, int Sseq) {
    const int lane = threadIdx.x & 31;
    const int wid  = threadIdx.x >> 5;
    const int half = lane >> 4;
    const int nl   = lane & 15;
    const int m0   = blockIdx.x * 16;
    const int n0   = (blockIdx.y * 8 + wid) * 64;
    if (n0 >= N) return;
    const int ntiles = N >> 4;

    v8f acc[4] = {};

    const __bf16* aptr  = A + (size_t)(m0 + nl) * K + half * 16;
    const __bf16* bbase = Bp + ((size_t)(n0 >> 4)) * 512 + (size_t)lane * 16;
    const size_t  bstep = (size_t)ntiles * 512;   // elements per k-step
    const int kloops = K >> 5;

    AF a0, a1;
    AF b0[4], b1[4];

    auto loadA = [&](AF& dst, int step) {
        const __bf16* p = aptr + (size_t)step * 32;
        dst.q[0] = *(const uint4*)(p);
        dst.q[1] = *(const uint4*)(p + 8);
    };
    auto loadB = [&](AF (&dst)[4], int step) {
        const __bf16* bk = bbase + (size_t)step * bstep;
#pragma unroll
        for (int j = 0; j < 4; ++j) {
            const __bf16* bp = bk + (size_t)j * 512;
            dst[j].q[0] = *(const uint4*)bp;
            dst[j].q[1] = *(const uint4*)(bp + 8);
        }
    };
    auto mma = [&](AF& a, AF (&b)[4]) {
#pragma unroll
        for (int j = 0; j < 4; ++j) {
            acc[j] = __builtin_amdgcn_wmma_f32_16x16x32_bf16(
                false, a.v, false, b[j].v, (short)0, acc[j], false, false);
        }
    };

    loadA(a0, 0);
    loadB(b0, 0);
    for (int kk = 0; kk < kloops; kk += 2) {
        loadA(a1, kk + 1);
        loadB(b1, kk + 1);
        mma(a0, b0);
        if (kk + 2 < kloops) {
            __builtin_prefetch(bbase + (size_t)(kk + 3) * bstep, 0, 1);
            loadA(a0, kk + 2);
            loadB(b0, kk + 2);
        }
        mma(a1, b1);
    }

#pragma unroll
    for (int j = 0; j < 4; ++j) {
        int n = n0 + j * 16 + nl;
        float bv = bias[n];
#pragma unroll
        for (int r = 0; r < 8; ++r) {
            int m = m0 + r + 8 * half;
            float v = acc[j][r] + bv;
            if (MODE == 0) {
                ((__bf16*)out)[(size_t)m * N + n] = (__bf16)v;
            } else if (MODE == 1) {
                int bidx = m / Sseq, s = m % Sseq;
                int nh = n >> 6, hd = n & 63;
                ((__bf16*)out)[(((size_t)bidx * NHEAD + nh) * HDIM + hd) * (size_t)Sseq + s] = (__bf16)v;
            } else {
                ((float*)out)[(size_t)m * N + n] = v + res[(size_t)m * N + n];
            }
        }
    }
}

// ---------------------------------------------------------------------------
// Fused gate/up GEMM: out = silu(h@Wg + bg) * (h@Wu + bu), bf16 [T, MLP]
// One wave computes a 16 x 32 strip for BOTH gate and up; same structural
// ping-pong as gemm_kernel (no runtime-indexed register arrays).
// ---------------------------------------------------------------------------
__global__ void gemm_gateup_kernel(const __bf16* __restrict__ A,
                                   const __bf16* __restrict__ Bg, const __bf16* __restrict__ Bu,
                                   const float* __restrict__ bg, const float* __restrict__ bu,
                                   __bf16* __restrict__ out, int T, int N, int K) {
    const int lane = threadIdx.x & 31;
    const int wid  = threadIdx.x >> 5;
    const int half = lane >> 4;
    const int nl   = lane & 15;
    const int m0   = blockIdx.x * 16;
    const int n0   = (blockIdx.y * 8 + wid) * 32;
    if (n0 >= N) return;
    const int ntiles = N >> 4;

    v8f accg[2] = {};
    v8f accu[2] = {};

    const __bf16* aptr  = A + (size_t)(m0 + nl) * K + half * 16;
    const size_t  fbase = ((size_t)(n0 >> 4)) * 512 + (size_t)lane * 16;
    const size_t  bstep = (size_t)ntiles * 512;
    const int kloops = K >> 5;

    AF a0, a1;
    AF g0[2], g1[2], u0[2], u1[2];

    auto loadA = [&](AF& dst, int step) {
        const __bf16* p = aptr + (size_t)step * 32;
        dst.q[0] = *(const uint4*)(p);
        dst.q[1] = *(const uint4*)(p + 8);
    };
    auto loadGU = [&](AF (&dg)[2], AF (&du)[2], int step) {
        const size_t fo = fbase + (size_t)step * bstep;
#pragma unroll
        for (int j = 0; j < 2; ++j) {
            const __bf16* gp = Bg + fo + (size_t)j * 512;
            const __bf16* up = Bu + fo + (size_t)j * 512;
            dg[j].q[0] = *(const uint4*)gp;
            dg[j].q[1] = *(const uint4*)(gp + 8);
            du[j].q[0] = *(const uint4*)up;
            du[j].q[1] = *(const uint4*)(up + 8);
        }
    };
    auto mma = [&](AF& a, AF (&bg_)[2], AF (&bu_)[2]) {
#pragma unroll
        for (int j = 0; j < 2; ++j) {
            accg[j] = __builtin_amdgcn_wmma_f32_16x16x32_bf16(
                false, a.v, false, bg_[j].v, (short)0, accg[j], false, false);
            accu[j] = __builtin_amdgcn_wmma_f32_16x16x32_bf16(
                false, a.v, false, bu_[j].v, (short)0, accu[j], false, false);
        }
    };

    loadA(a0, 0);
    loadGU(g0, u0, 0);
    for (int kk = 0; kk < kloops; kk += 2) {
        loadA(a1, kk + 1);
        loadGU(g1, u1, kk + 1);
        mma(a0, g0, u0);
        if (kk + 2 < kloops) {
            __builtin_prefetch(Bg + fbase + (size_t)(kk + 3) * bstep, 0, 1);
            __builtin_prefetch(Bu + fbase + (size_t)(kk + 3) * bstep, 0, 1);
            loadA(a0, kk + 2);
            loadGU(g0, u0, kk + 2);
        }
        mma(a1, g1, u1);
    }

#pragma unroll
    for (int j = 0; j < 2; ++j) {
        int n = n0 + j * 16 + nl;
        float bgv = bg[n], buv = bu[n];
#pragma unroll
        for (int r = 0; r < 8; ++r) {
            int m = m0 + r + 8 * half;
            float g = accg[j][r] + bgv;
            float u = accu[j][r] + buv;
            float s = g / (1.f + __expf(-g));
            out[(size_t)m * N + n] = (__bf16)(s * u);
        }
    }
}

// ---------------------------------------------------------------------------
// Flash attention (causal), one wave per 16-query tile per (b, head).
// Q,K : bf16 [b, s, nh, hd] row-major;  Vt : bf16 [b, nh, hd, s]
// O   : bf16 [b, s, nh, hd]
// ---------------------------------------------------------------------------
__global__ void attn_kernel(const __bf16* __restrict__ Q, const __bf16* __restrict__ Kb,
                            const __bf16* __restrict__ Vt, __bf16* __restrict__ O,
                            int Sseq, int D) {
    __shared__ __align__(16) __bf16 pbuf[8][16 * 40];
    const int lane = threadIdx.x & 31;
    const int wid  = threadIdx.x >> 5;
    const int half = lane >> 4;
    const int nl   = lane & 15;
    const int head = blockIdx.x & (NHEAD - 1);
    const int b    = blockIdx.x >> 4;
    const int s0   = (blockIdx.y * 8 + wid) * 16;
    const size_t tokbase = (size_t)b * Sseq;

    // Q A-fragments (hd chunks of 32), reused all iterations
    const __bf16* qbase = Q + (tokbase + s0) * (size_t)D + head * HDIM;
    AF qf0, qf1;
    {
        const __bf16* p = qbase + (size_t)nl * D + half * 16;
        qf0.q[0] = *(const uint4*)p;
        qf0.q[1] = *(const uint4*)(p + 8);
        qf1.q[0] = *(const uint4*)(p + 32);
        qf1.q[1] = *(const uint4*)(p + 40);
    }

    v8f oacc[4] = {};
    float mrow[8], lsum[8];
#pragma unroll
    for (int r = 0; r < 8; ++r) { mrow[r] = -1e30f; lsum[r] = 0.f; }

    __bf16* pl = pbuf[wid];
    const int nchunk = (s0 + 47) >> 5;   // chunks of 32 keys covering t <= s0+15

    for (int c = 0; c < nchunk; ++c) {
        const int t0 = c * 32;

        // ---- load all four K B-fragments first, then 4 WMMAs (overlap) ----
        AF bkf[2][2];
#pragma unroll
        for (int j = 0; j < 2; ++j) {
#pragma unroll
            for (int kk = 0; kk < 2; ++kk) {
                const __bf16* kp = Kb + (tokbase + t0 + j * 16 + nl) * (size_t)D
                                      + head * HDIM + kk * 32 + half * 16;
                bkf[j][kk].q[0] = *(const uint4*)kp;
                bkf[j][kk].q[1] = *(const uint4*)(kp + 8);
            }
        }
        // and prefetch V for this chunk while logits compute
        {
            const __bf16* vp0 = Vt + (((size_t)b * NHEAD + head) * HDIM + nl) * (size_t)Sseq + t0;
            __builtin_prefetch(vp0, 0, 1);
        }

        v8f lg[2] = {};
#pragma unroll
        for (int j = 0; j < 2; ++j) {
            lg[j] = __builtin_amdgcn_wmma_f32_16x16x32_bf16(
                false, qf0.v, false, bkf[j][0].v, (short)0, lg[j], false, false);
            lg[j] = __builtin_amdgcn_wmma_f32_16x16x32_bf16(
                false, qf1.v, false, bkf[j][1].v, (short)0, lg[j], false, false);
        }

        // scale + causal mask
#pragma unroll
        for (int r = 0; r < 8; ++r) {
            int srow = s0 + r + 8 * half;
#pragma unroll
            for (int j = 0; j < 2; ++j) {
                int t = t0 + j * 16 + nl;
                float v = lg[j][r] * 0.125f;
                lg[j][r] = (t > srow) ? -1e30f : v;
            }
        }

        // online softmax: row reductions across the 16-lane groups
        float mnew[8], corr[8];
#pragma unroll
        for (int r = 0; r < 8; ++r) {
            float lm = fmaxf(lg[0][r], lg[1][r]);
            for (int off = 1; off < 16; off <<= 1)
                lm = fmaxf(lm, __shfl_xor(lm, off, 32));
            mnew[r] = fmaxf(mrow[r], lm);
            corr[r] = __expf(mrow[r] - mnew[r]);
            mrow[r] = mnew[r];
        }
#pragma unroll
        for (int r = 0; r < 8; ++r) {
            float p0 = __expf(lg[0][r] - mnew[r]);
            float p1 = __expf(lg[1][r] - mnew[r]);
            float rs = p0 + p1;
            for (int off = 1; off < 16; off <<= 1)
                rs += __shfl_xor(rs, off, 32);
            lsum[r] = lsum[r] * corr[r] + rs;
#pragma unroll
            for (int jj = 0; jj < 4; ++jj) oacc[jj][r] *= corr[r];
            // scatter P into LDS [m][t_local] for the A-fragment transpose
            int m = r + 8 * half;
            pl[m * 40 + nl]      = (__bf16)p0;
            pl[m * 40 + 16 + nl] = (__bf16)p1;
        }

        // ---- load all four V B-fragments while LDS drains ----
        AF bvf[4];
#pragma unroll
        for (int jj = 0; jj < 4; ++jj) {
            const __bf16* vp = Vt + (((size_t)b * NHEAD + head) * HDIM + jj * 16 + nl) * (size_t)Sseq
                                  + t0 + half * 16;
            bvf[jj].q[0] = *(const uint4*)vp;
            bvf[jj].q[1] = *(const uint4*)(vp + 8);
        }

        __asm__ volatile("s_wait_dscnt 0x0" ::: "memory");

        // reload P as A-fragment (16 x 32)
        AF pa;
        {
            const __bf16* pp = pl + (size_t)nl * 40 + half * 16;
            pa.q[0] = *(const uint4*)pp;
            pa.q[1] = *(const uint4*)(pp + 8);
        }

        // O += P @ V_chunk
#pragma unroll
        for (int jj = 0; jj < 4; ++jj) {
            oacc[jj] = __builtin_amdgcn_wmma_f32_16x16x32_bf16(
                false, pa.v, false, bvf[jj].v, (short)0, oacc[jj], false, false);
        }
    }

    // normalize and write attn_out (bf16, row-major [tok, D])
#pragma unroll
    for (int jj = 0; jj < 4; ++jj) {
#pragma unroll
        for (int r = 0; r < 8; ++r) {
            int s  = s0 + r + 8 * half;
            int hd = jj * 16 + nl;
            O[(tokbase + s) * (size_t)D + head * HDIM + hd] = (__bf16)(oacc[jj][r] / lsum[r]);
        }
    }
}

// ---------------------------------------------------------------------------
// Host orchestration
// ---------------------------------------------------------------------------
extern "C" void kernel_launch(void* const* d_in, const int* in_sizes, int n_in,
                              void* d_out, int out_size, void* d_ws, size_t ws_size,
                              hipStream_t stream) {
    (void)in_sizes; (void)n_in; (void)out_size; (void)ws_size;
    const int Bb = 2, S = 2048, D = 1024, MLP = 4096;
    const int T = Bb * S;

    const float* x        = (const float*)d_in[0];
    const float* attn_ns  = (const float*)d_in[2];
    const float* q_w      = (const float*)d_in[3];
    const float* q_b      = (const float*)d_in[4];
    const float* k_w      = (const float*)d_in[5];
    const float* k_b      = (const float*)d_in[6];
    const float* v_w      = (const float*)d_in[7];
    const float* v_b      = (const float*)d_in[8];
    const float* o_w      = (const float*)d_in[9];
    const float* o_b      = (const float*)d_in[10];
    const float* ffn_ns   = (const float*)d_in[11];
    const float* gate_w   = (const float*)d_in[12];
    const float* gate_b   = (const float*)d_in[13];
    const float* up_w     = (const float*)d_in[14];
    const float* up_b     = (const float*)d_in[15];
    const float* down_w   = (const float*)d_in[16];
    const float* down_b   = (const float*)d_in[17];
    float* out = (float*)d_out;

    // workspace carve-out
    char* ws = (char*)d_ws;
    size_t off = 0;
    auto carve = [&](size_t bytes) -> void* {
        void* p = ws + off;
        off += (bytes + 255) & ~(size_t)255;
        return p;
    };
    __bf16* h1   = (__bf16*)carve((size_t)T * D * 2);
    __bf16* qb   = (__bf16*)carve((size_t)T * D * 2);
    __bf16* kb   = (__bf16*)carve((size_t)T * D * 2);
    __bf16* vtb  = (__bf16*)carve((size_t)T * D * 2);
    __bf16* ao   = (__bf16*)carve((size_t)T * D * 2);
    float*  x2   = (float*)carve((size_t)T * D * 4);
    __bf16* h2   = (__bf16*)carve((size_t)T * D * 2);
    __bf16* mid  = (__bf16*)carve((size_t)T * MLP * 2);
    __bf16* pq   = (__bf16*)carve((size_t)D * D * 2);
    __bf16* pk   = (__bf16*)carve((size_t)D * D * 2);
    __bf16* pv   = (__bf16*)carve((size_t)D * D * 2);
    __bf16* po   = (__bf16*)carve((size_t)D * D * 2);
    __bf16* pg   = (__bf16*)carve((size_t)D * MLP * 2);
    __bf16* pu   = (__bf16*)carve((size_t)D * MLP * 2);
    __bf16* pd   = (__bf16*)carve((size_t)MLP * D * 2);

    // 1) prepack weights into WMMA B-fragment layout (bf16)
    {
        int thrDD  = D * D / 16;
        int thrDM  = D * MLP / 16;
        pack_w_kernel<<<(thrDD + 255) / 256, 256, 0, stream>>>(q_w, pq, D, D);
        pack_w_kernel<<<(thrDD + 255) / 256, 256, 0, stream>>>(k_w, pk, D, D);
        pack_w_kernel<<<(thrDD + 255) / 256, 256, 0, stream>>>(v_w, pv, D, D);
        pack_w_kernel<<<(thrDD + 255) / 256, 256, 0, stream>>>(o_w, po, D, D);
        pack_w_kernel<<<(thrDM + 255) / 256, 256, 0, stream>>>(gate_w, pg, D, MLP);
        pack_w_kernel<<<(thrDM + 255) / 256, 256, 0, stream>>>(up_w, pu, D, MLP);
        pack_w_kernel<<<(thrDM + 255) / 256, 256, 0, stream>>>(down_w, pd, MLP, D);
    }

    // 2) attention RMSNorm
    rmsnorm_kernel<<<T, 256, 0, stream>>>(x, attn_ns, h1, D);

    // 3) Q/K/V projections
    {
        dim3 g(T / 16, D / (64 * 8));
        gemm_kernel<0><<<g, 256, 0, stream>>>(h1, pq, q_b, nullptr, qb, T, D, D, S);
        gemm_kernel<0><<<g, 256, 0, stream>>>(h1, pk, k_b, nullptr, kb, T, D, D, S);
        gemm_kernel<1><<<g, 256, 0, stream>>>(h1, pv, v_b, nullptr, vtb, T, D, D, S);
    }

    // 4) causal flash attention
    {
        dim3 g(Bb * NHEAD, S / (16 * 8));
        attn_kernel<<<g, 256, 0, stream>>>(qb, kb, vtb, ao, S, D);
    }

    // 5) output projection + residual -> x2 (fp32)
    {
        dim3 g(T / 16, D / (64 * 8));
        gemm_kernel<2><<<g, 256, 0, stream>>>(ao, po, o_b, x, x2, T, D, D, S);
    }

    // 6) FFN RMSNorm
    rmsnorm_kernel<<<T, 256, 0, stream>>>(x2, ffn_ns, h2, D);

    // 7) fused gate/up + SiLU
    {
        dim3 g(T / 16, MLP / (32 * 8));
        gemm_gateup_kernel<<<g, 256, 0, stream>>>(h2, pg, pu, gate_b, up_b, mid, T, MLP, D);
    }

    // 8) down projection + residual -> d_out (fp32)
    {
        dim3 g(T / 16, D / (64 * 8));
        gemm_kernel<2><<<g, 256, 0, stream>>>(mid, pd, down_b, x2, out, T, D, MLP, S);
    }
}
